// RNNStyleTransformer_32847909880456
// MI455X (gfx1250) — matrix operand order repaired and checked by reading, not compiled, not run on hardware
//
#include <hip/hip_runtime.h>
#include <hip/hip_bf16.h>

#define T_STEPS 64
#define BATCH   32
#define IN_DIM  256
#define HIST    32
#define SEQ     33
#define D_MODEL 1024
#define NHEAD   16
#define DHEAD   64
#define FF_DIM  4096
#define NLAYER  4
#define CHSEQ   256                 // sequences per chunk (divides 2048)
#define MC      (CHSEQ * SEQ)       // 8448 rows per chunk, divisible by 64

typedef unsigned short u16;
typedef __attribute__((ext_vector_type(16))) __bf16 v16bf;
typedef __attribute__((ext_vector_type(8)))  float  v8f;

union ABf { uint4 q[2]; v16bf v; };

__device__ __forceinline__ u16 f2bf(float f) {
  unsigned u = __float_as_uint(f);
  u += 0x7FFFu + ((u >> 16) & 1u);       // round-to-nearest-even
  return (u16)(u >> 16);
}
__device__ __forceinline__ float bf2f(u16 h) {
  return __uint_as_float(((unsigned)h) << 16);
}

// ---------------- f32 -> bf16 conversion (weights) ----------------
__global__ void conv_bf16_kernel(const float* __restrict__ src,
                                 u16* __restrict__ dst, int n) {
  int i = blockIdx.x * 256 + threadIdx.x;
  if (i < n) dst[i] = f2bf(src[i]);
}

__global__ void copy_f32_kernel(const float* __restrict__ src,
                                float* __restrict__ dst, int n) {
  int i = blockIdx.x * 256 + threadIdx.x;
  if (i < n) dst[i] = src[i];
}

// ---------------- gather sliding windows -> bf16 ----------------
__global__ void gather_kernel(const float* __restrict__ x,
                              const float* __restrict__ hidden,
                              u16* __restrict__ seqin, int chunk) {
  int r  = blockIdx.x;
  int si = r / SEQ, s = r % SEQ;
  int g  = chunk * CHSEQ + si;
  int t  = g / BATCH, b = g % BATCH;
  int tok = t + s;
  const float* src = (tok < HIST)
      ? (hidden + ((size_t)tok * BATCH + b) * IN_DIM)
      : (x + ((size_t)(tok - HIST) * BATCH + b) * IN_DIM);
  seqin[(size_t)r * IN_DIM + threadIdx.x] = f2bf(src[threadIdx.x]);
}

// ---------------- bf16 WMMA GEMM ----------------
// out = A(MxK) * Bw(KxN) + bias [+resid][relu]
// 256 threads = 8 waves; WG tile 64(M) x 256(N); wave tile 32 x 64.
// A tile (64x32 bf16) double-buffered in LDS via async global->LDS copies.
__global__ __launch_bounds__(256) void gemm_bf16(
    const u16* __restrict__ A, const u16* __restrict__ Bw,
    const float* __restrict__ bias, const float* __restrict__ resid,
    float* __restrict__ outF, u16* __restrict__ outB,
    int M, int N, int K, int relu)
{
  __shared__ __align__(16) u16 As[2][64][40];   // 80B row stride (16B multiple)
  const int tid  = threadIdx.x;
  const int wave = tid >> 5, lane = tid & 31;
  const int mt = wave >> 2, nq = wave & 3;      // mt: 0..1 (32 rows), nq: 0..3 (64 cols)
  const int m0 = blockIdx.x * 64;
  const int n0 = blockIdx.y * 256 + nq * 64;

  v8f acc[2][4];
  #pragma unroll
  for (int ms = 0; ms < 2; ms++)
    #pragma unroll
    for (int ns = 0; ns < 4; ns++)
      #pragma unroll
      for (int e = 0; e < 8; e++) acc[ms][ns][e] = 0.0f;

  // async staging: each of 256 threads copies one 16B chunk (64 rows x 64B)
  const int srow = tid >> 2;             // 0..63
  const int schk = (tid & 3) * 8;        // element offset 0,8,16,24
  const u16* abase = A + (size_t)(m0 + srow) * K + schk;

  const int row16 = lane & 15;
  const int kb = (lane < 16) ? 0 : 8;    // ISA 16-bit A layout K grouping

  const int nIter = K / 32;

  // prologue: stage tile 0 into buffer 0
  {
    unsigned ldsoff = (unsigned)(size_t)&As[0][srow][schk];
    const u16* gsrc = abase;
    asm volatile("global_load_async_to_lds_b128 %0, %1, off"
                 :: "v"(ldsoff), "v"(gsrc) : "memory");
  }

  for (int it = 0; it < nIter; it++) {
    const int cur = it & 1;
    asm volatile("s_wait_asynccnt 0x0" ::: "memory");
    __syncthreads();
    if (it + 1 < nIter) {                 // stage next tile into other buffer
      unsigned ldsoff = (unsigned)(size_t)&As[cur ^ 1][srow][schk];
      const u16* gsrc = abase + (it + 1) * 32;
      asm volatile("global_load_async_to_lds_b128 %0, %1, off"
                   :: "v"(ldsoff), "v"(gsrc) : "memory");
    }

    ABf a[2];
    #pragma unroll
    for (int ms = 0; ms < 2; ms++) {
      const int r = mt * 32 + ms * 16 + row16;
      a[ms].q[0] = *(const uint4*)&As[cur][r][kb];       // K kb..kb+7
      a[ms].q[1] = *(const uint4*)&As[cur][r][kb + 16];  // K kb+16..kb+23
    }

    const u16* bp = Bw + (size_t)(it * 32) * N + n0 + row16;   // lane = N column
    #pragma unroll
    for (int ns = 0; ns < 4; ns++) {
      ABf b;
      #pragma unroll
      for (int s = 0; s < 16; s++) {
        const int kk = (s < 8) ? (kb + s) : (16 + kb + s - 8);
        b.v[s] = __builtin_bit_cast(__bf16, bp[(size_t)kk * N + ns * 16]);
      }
      #pragma unroll
      for (int ms = 0; ms < 2; ms++)
        acc[ms][ns] = __builtin_amdgcn_wmma_f32_16x16x32_bf16(
            false, a[ms].v, false, b.v, (short)0, acc[ms][ns], false, false);
    }
  }

  // C/D layout: VGPR e: lanes 0-15 -> M=e, lanes 16-31 -> M=8+e; N = lane%16
  const int mloc = ((lane < 16) ? 0 : 8);
  const int ncol = lane & 15;
  #pragma unroll
  for (int ms = 0; ms < 2; ms++) {
    #pragma unroll
    for (int ns = 0; ns < 4; ns++) {
      #pragma unroll
      for (int e = 0; e < 8; e++) {
        int m = m0 + mt * 32 + ms * 16 + mloc + e;
        int n = n0 + ns * 16 + ncol;
        size_t idx = (size_t)m * N + n;
        float v = acc[ms][ns][e] + bias[n];
        if (resid) v += resid[idx];
        if (relu && v < 0.0f) v = 0.0f;
        if (outF) outF[idx] = v;
        if (outB) outB[idx] = f2bf(v);
      }
    }
  }
}

// ---------------- attention: one block per (sequence, head) ----------------
__global__ __launch_bounds__(256) void attn_kernel(
    const u16* __restrict__ qb, const u16* __restrict__ kbuf,
    const u16* __restrict__ vb, u16* __restrict__ attnb)
{
  __shared__ float qs[SEQ][DHEAD];
  __shared__ float ks[SEQ][DHEAD];
  __shared__ float vs[SEQ][DHEAD];
  __shared__ float pr[8][40];
  const int si = blockIdx.x / NHEAD;
  const int hh = blockIdx.x % NHEAD;
  const int tid = threadIdx.x;
  const size_t rbase = (size_t)si * SEQ;

  for (int i = tid; i < SEQ * DHEAD; i += 256) {
    int s = i / DHEAD, d = i % DHEAD;
    size_t idx = (rbase + s) * D_MODEL + (size_t)hh * DHEAD + d;
    qs[s][d] = bf2f(qb[idx]);
    ks[s][d] = bf2f(kbuf[idx]);
    vs[s][d] = bf2f(vb[idx]);
  }
  __syncthreads();

  const int wave = tid >> 5, lane = tid & 31;
  for (int it = 0; it < 5; it++) {
    int qi = it * 8 + wave;
    if (qi < SEQ) {
      float s1 = 0.f, s2 = 0.f;
      #pragma unroll 8
      for (int d = 0; d < DHEAD; d++) s1 += qs[qi][d] * ks[lane][d];
      int ki2 = lane + 32;
      if (ki2 < SEQ)
        for (int d = 0; d < DHEAD; d++) s2 += qs[qi][d] * ks[ki2][d];
      s1 *= 0.125f; s2 *= 0.125f;          // 1/sqrt(64)
      bool v1 = (lane <= qi);
      bool v2 = (ki2 < SEQ) && (ki2 <= qi);
      float mx = v1 ? s1 : -1e30f;
      if (v2 && s2 > mx) mx = s2;
      for (int off = 16; off > 0; off >>= 1) {
        float o = __shfl_xor(mx, off, 32);
        if (o > mx) mx = o;
      }
      float e1 = v1 ? __expf(s1 - mx) : 0.f;
      float e2 = v2 ? __expf(s2 - mx) : 0.f;
      float sum = e1 + e2;
      for (int off = 16; off > 0; off >>= 1) sum += __shfl_xor(sum, off, 32);
      float inv = 1.0f / sum;
      pr[wave][lane] = e1 * inv;
      if (lane == 0) pr[wave][32] = e2 * inv;  // same-wave LDS, in-order

      float o0 = 0.f, o1 = 0.f;
      int nk = qi + 1;
      for (int ki = 0; ki < nk; ki++) {
        float a = pr[wave][ki];
        o0 += a * vs[ki][lane];
        o1 += a * vs[ki][lane + 32];
      }
      size_t obase = (rbase + qi) * D_MODEL + (size_t)hh * DHEAD;
      attnb[obase + lane]      = f2bf(o0);
      attnb[obase + lane + 32] = f2bf(o1);
    }
  }
}

// ---------------- LayerNorm over D=1024, one block per row ----------------
__global__ __launch_bounds__(256) void ln_kernel(
    const float* __restrict__ in, const float* __restrict__ g,
    const float* __restrict__ b, float* __restrict__ outF,
    u16* __restrict__ outB)
{
  __shared__ float red[2][8];
  const int r = blockIdx.x;
  const int tid = threadIdx.x;
  const float* row = in + (size_t)r * D_MODEL;
  float x[4], s = 0.f, ss = 0.f;
  #pragma unroll
  for (int i = 0; i < 4; i++) {
    x[i] = row[tid + i * 256];
    s += x[i]; ss += x[i] * x[i];
  }
  for (int off = 16; off > 0; off >>= 1) {
    s  += __shfl_xor(s,  off, 32);
    ss += __shfl_xor(ss, off, 32);
  }
  int wave = tid >> 5, lane = tid & 31;
  if (lane == 0) { red[0][wave] = s; red[1][wave] = ss; }
  __syncthreads();
  s = 0.f; ss = 0.f;
  #pragma unroll
  for (int w = 0; w < 8; w++) { s += red[0][w]; ss += red[1][w]; }
  float mean = s * (1.0f / D_MODEL);
  float var  = ss * (1.0f / D_MODEL) - mean * mean;
  float rstd = __frsqrt_rn(var + 1e-5f);
  #pragma unroll
  for (int i = 0; i < 4; i++) {
    int c = tid + i * 256;
    float y = (x[i] - mean) * rstd * g[c] + b[c];
    outF[(size_t)r * D_MODEL + c] = y;
    outB[(size_t)r * D_MODEL + c] = f2bf(y);
  }
}

// ---------------- scatter last-token rows to outs ----------------
__global__ void out_kernel(const float* __restrict__ h,
                           float* __restrict__ out, int chunk) {
  int si = blockIdx.x;
  int g  = chunk * CHSEQ + si;               // g = t*B + b
  size_t src = ((size_t)si * SEQ + (SEQ - 1)) * D_MODEL;
  size_t dst = (size_t)g * D_MODEL;
  for (int i = threadIdx.x; i < D_MODEL; i += 256)
    out[dst + i] = h[src + i];
}

extern "C" void kernel_launch(void* const* d_in, const int* in_sizes, int n_in,
                              void* d_out, int out_size, void* d_ws, size_t ws_size,
                              hipStream_t stream) {
  (void)in_sizes; (void)n_in; (void)out_size; (void)ws_size;
  const float* x      = (const float*)d_in[0];
  const float* hidden = (const float*)d_in[1];
  const float* proj_w = (const float*)d_in[2];
  const float* proj_b = (const float*)d_in[3];
  const float* Wq = (const float*)d_in[4];
  const float* bq = (const float*)d_in[5];
  const float* Wk = (const float*)d_in[6];
  const float* bk = (const float*)d_in[7];
  const float* Wv = (const float*)d_in[8];
  const float* bv = (const float*)d_in[9];
  const float* Wo = (const float*)d_in[10];
  const float* bo = (const float*)d_in[11];
  const float* ln1_g = (const float*)d_in[12];
  const float* ln1_b = (const float*)d_in[13];
  const float* ln2_g = (const float*)d_in[14];
  const float* ln2_b = (const float*)d_in[15];
  const float* W1 = (const float*)d_in[16];
  const float* b1 = (const float*)d_in[17];
  const float* W2 = (const float*)d_in[18];
  const float* b2 = (const float*)d_in[19];
  float* out = (float*)d_out;

  char* ws = (char*)d_ws;
  size_t off = 0;
  auto alloc = [&](size_t bytes) -> void* {
    void* p = ws + off;
    off += (bytes + 255) & ~(size_t)255;
    return p;
  };

  const size_t DD  = (size_t)D_MODEL * D_MODEL;
  const size_t DF  = (size_t)D_MODEL * FF_DIM;
  u16* wb_proj = (u16*)alloc((size_t)IN_DIM * D_MODEL * 2);
  u16* wb_q = (u16*)alloc(NLAYER * DD * 2);
  u16* wb_k = (u16*)alloc(NLAYER * DD * 2);
  u16* wb_v = (u16*)alloc(NLAYER * DD * 2);
  u16* wb_o = (u16*)alloc(NLAYER * DD * 2);
  u16* wb_1 = (u16*)alloc(NLAYER * DF * 2);
  u16* wb_2 = (u16*)alloc(NLAYER * DF * 2);
  u16* seqin = (u16*)alloc((size_t)MC * IN_DIM * 2);
  u16* hb    = (u16*)alloc((size_t)MC * D_MODEL * 2);
  u16* qbuf  = (u16*)alloc((size_t)MC * D_MODEL * 2);
  u16* kbuf  = (u16*)alloc((size_t)MC * D_MODEL * 2);
  u16* vbuf  = (u16*)alloc((size_t)MC * D_MODEL * 2);
  u16* abuf  = (u16*)alloc((size_t)MC * D_MODEL * 2);
  u16* f1    = (u16*)alloc((size_t)MC * FF_DIM * 2);
  float* h   = (float*)alloc((size_t)MC * D_MODEL * 4);
  float* tmp = (float*)alloc((size_t)MC * D_MODEL * 4);

  auto conv = [&](const float* s, u16* d, size_t n) {
    conv_bf16_kernel<<<dim3((unsigned)((n + 255) / 256)), 256, 0, stream>>>(s, d, (int)n);
  };
  conv(proj_w, wb_proj, (size_t)IN_DIM * D_MODEL);
  conv(Wq, wb_q, NLAYER * DD);
  conv(Wk, wb_k, NLAYER * DD);
  conv(Wv, wb_v, NLAYER * DD);
  conv(Wo, wb_o, NLAYER * DD);
  conv(W1, wb_1, NLAYER * DF);
  conv(W2, wb_2, NLAYER * DF);

  const int NCHUNK = (T_STEPS * BATCH) / CHSEQ;   // 8
  for (int chunk = 0; chunk < NCHUNK; chunk++) {
    gather_kernel<<<MC, 256, 0, stream>>>(x, hidden, seqin, chunk);
    // h0 = seqin @ proj_w + proj_b  (f32 + bf16 copies)
    gemm_bf16<<<dim3(MC / 64, D_MODEL / 256), 256, 0, stream>>>(
        seqin, wb_proj, proj_b, nullptr, h, hb, MC, D_MODEL, IN_DIM, 0);

    for (int l = 0; l < NLAYER; l++) {
      gemm_bf16<<<dim3(MC / 64, D_MODEL / 256), 256, 0, stream>>>(
          hb, wb_q + (size_t)l * DD, bq + (size_t)l * D_MODEL, nullptr,
          nullptr, qbuf, MC, D_MODEL, D_MODEL, 0);
      gemm_bf16<<<dim3(MC / 64, D_MODEL / 256), 256, 0, stream>>>(
          hb, wb_k + (size_t)l * DD, bk + (size_t)l * D_MODEL, nullptr,
          nullptr, kbuf, MC, D_MODEL, D_MODEL, 0);
      gemm_bf16<<<dim3(MC / 64, D_MODEL / 256), 256, 0, stream>>>(
          hb, wb_v + (size_t)l * DD, bv + (size_t)l * D_MODEL, nullptr,
          nullptr, vbuf, MC, D_MODEL, D_MODEL, 0);
      attn_kernel<<<CHSEQ * NHEAD, 256, 0, stream>>>(qbuf, kbuf, vbuf, abuf);
      // tmp = h + attn @ Wo + bo
      gemm_bf16<<<dim3(MC / 64, D_MODEL / 256), 256, 0, stream>>>(
          abuf, wb_o + (size_t)l * DD, bo + (size_t)l * D_MODEL, h,
          tmp, nullptr, MC, D_MODEL, D_MODEL, 0);
      ln_kernel<<<MC, 256, 0, stream>>>(
          tmp, ln1_g + (size_t)l * D_MODEL, ln1_b + (size_t)l * D_MODEL, h, hb);
      // f1 = relu(h @ W1 + b1)  (bf16)
      gemm_bf16<<<dim3(MC / 64, FF_DIM / 256), 256, 0, stream>>>(
          hb, wb_1 + (size_t)l * DF, b1 + (size_t)l * FF_DIM, nullptr,
          nullptr, f1, MC, FF_DIM, D_MODEL, 1);
      // tmp = h + f1 @ W2 + b2
      gemm_bf16<<<dim3(MC / 64, D_MODEL / 256), 256, 0, stream>>>(
          f1, wb_2 + (size_t)l * DF, b2 + (size_t)l * D_MODEL, h,
          tmp, nullptr, MC, D_MODEL, FF_DIM, 0);
      ln_kernel<<<MC, 256, 0, stream>>>(
          tmp, ln2_g + (size_t)l * D_MODEL, ln2_b + (size_t)l * D_MODEL, h, hb);
    }
    out_kernel<<<CHSEQ, 256, 0, stream>>>(h, out, chunk);
  }

  // h_fin (transposed back) == x[32:64], contiguous copy
  const int nfin = HIST * BATCH * IN_DIM;
  copy_f32_kernel<<<(nfin + 255) / 256, 256, 0, stream>>>(
      x + (size_t)HIST * BATCH * IN_DIM,
      out + (size_t)T_STEPS * BATCH * D_MODEL, nfin);
}
